// BPRLoss_21045339750999
// MI455X (gfx1250) — compile-verified, matrix operand order checked by reading, not run on hardware
//
#include <hip/hip_runtime.h>
#include <hip/hip_bf16.h>

typedef __attribute__((ext_vector_type(16))) _Float16 v16h;
typedef __attribute__((ext_vector_type(8)))  float    v8f;

#define N_TOTAL      16777216
#define NUM_POS      65536
#define NUM_NEGS     (N_TOTAL - NUM_POS)          /* 16711680 = 255 * 65536 */
#define NUM_NEG_S    16
#define NUM_SAMPLES  (NUM_POS * NUM_NEG_S)        /* 1048576 */

#define LEAF            128
#define LEAVES_PER_SUP  512                       /* superblock = 65536 elems */
#define NUM_LEAVES      (NUM_NEGS / LEAF)         /* 130560 */
#define NUM_SUPER       255

/* workspace layout (bytes) */
#define WS_MIN       0u
#define WS_SUPERCUM  64u                          /* 256 floats (total at [255]) */
#define WS_LEAFSUM   4096u                        /* 130560 floats */
#define WS_LEAFCUM   (4096u + 130560u * 4u)       /* 130560 floats -> ends 1048576 */
#define WS_PARTIAL   1048576u                     /* 2048 floats (per-wave partials) */
#define WS_NEEDED    (WS_PARTIAL + 2048u * 4u)

/* ---- order-preserving float<->uint encoding for atomicMin over f32 ---- */
__device__ __forceinline__ unsigned fenc(float f) {
    unsigned u = __float_as_uint(f);
    return (u & 0x80000000u) ? ~u : (u | 0x80000000u);
}
__device__ __forceinline__ float fdec(unsigned e) {
    unsigned u = (e & 0x80000000u) ? (e & 0x7FFFFFFFu) : ~e;
    return __uint_as_float(u);
}

/* deterministic counter-based RNG: splitmix64 -> 24-bit uniform in [0,1) */
__device__ __forceinline__ float rng_uniform(unsigned sid) {
    unsigned long long z = (unsigned long long)sid * 0x9E3779B97F4A7C15ull + 42ull;
    z = (z ^ (z >> 30)) * 0xBF58476D1CE4E5B9ull;
    z = (z ^ (z >> 27)) * 0x94D049BB133111EBull;
    z = z ^ (z >> 31);
    return (float)(unsigned)(z >> 40) * (1.0f / 16777216.0f);
}

/* =====================================================================
 * Pass 1: streaming reduction over negatives.
 * 510 blocks x 256 thr = 4080 waves; each wave owns 32 leaves of 128.
 * Per leaf: coalesced float4 load per lane, butterfly sum -> leafsum,
 * running min -> global atomicMin (order-encoded).
 * ===================================================================== */
__global__ __launch_bounds__(256)
void bpr_pass1(const float* __restrict__ out,
               unsigned* __restrict__ gmin,
               float* __restrict__ leafsum) {
    const int lane = threadIdx.x & 31;
    const int wid  = (int)((blockIdx.x * blockDim.x + threadIdx.x) >> 5);
    const float* negbase = out + NUM_POS;
    float rmin = 3.4028235e38f;

    #pragma unroll 1
    for (int it = 0; it < 32; ++it) {
        const int leaf = wid * 32 + it;
        const float* p = negbase + (size_t)leaf * LEAF + lane * 4;
        if (it < 31)
            __builtin_prefetch(p + LEAF, 0, 0);       /* global_prefetch_b8 */
        const float4 x = *(const float4*)p;
        float s = (x.x + x.y) + (x.z + x.w);
        rmin = fminf(rmin, fminf(fminf(x.x, x.y), fminf(x.z, x.w)));
        #pragma unroll
        for (int off = 16; off; off >>= 1)
            s += __shfl_xor(s, off, 32);
        if (lane == 0) leafsum[leaf] = s;
    }
    #pragma unroll
    for (int off = 16; off; off >>= 1)
        rmin = fminf(rmin, __shfl_xor(rmin, off, 32));
    if (lane == 0) atomicMin(gmin, fenc(rmin));
}

/* =====================================================================
 * Pass 2: build the 2-level CDF. Single block.
 * Thread s (< 255): sequential inclusive cumsum of 512 leaf w-sums
 * (w-sum = leafsum - 128*min) -> leafcum; superblock total -> supercum[s].
 * Thread 0 then cumsums the 255 superblock totals; total at [255].
 * ===================================================================== */
__global__ __launch_bounds__(256)
void bpr_scan(const unsigned* __restrict__ gmin,
              const float* __restrict__ leafsum,
              float* __restrict__ leafcum,
              float* __restrict__ supercum) {
    const float m = fdec(*gmin);
    const int s = threadIdx.x;
    if (s < NUM_SUPER) {
        float acc = 0.0f;
        const int base = s * LEAVES_PER_SUP;
        for (int j = 0; j < LEAVES_PER_SUP; ++j) {
            acc += leafsum[base + j] - (float)LEAF * m;
            leafcum[base + j] = acc;
        }
        supercum[s] = acc;
    }
    __syncthreads();
    if (threadIdx.x == 0) {
        float acc = 0.0f;
        for (int i = 0; i < NUM_SUPER; ++i) { acc += supercum[i]; supercum[i] = acc; }
        supercum[NUM_SUPER] = acc;           /* grand total of w */
    }
}

/* =====================================================================
 * Pass 3: weighted sampling + log-sigmoid terms + WMMA wave reduction.
 * 256 blocks x 256 thr = 65536 threads; thread t owns samples 16t..16t+15.
 * After the loop each wave holds 512 loss terms (16 per lane) -> exactly
 * one 16x32 f16 B-operand. A = ones, so D[m][n] = column sums; two WMMAs
 * (hi/lo f16 split, f32 accumulate) reduce all 512 terms. sum(c[0]) over
 * the 32 lanes = 2 * wave total. One partial per wave, no float atomics.
 * ===================================================================== */
__global__ __launch_bounds__(256)
void bpr_sample_loss(const float* __restrict__ out,
                     const unsigned* __restrict__ gmin,
                     const float* __restrict__ supercum,
                     const float* __restrict__ leafcum,
                     float* __restrict__ partial) {
    const unsigned tid = blockIdx.x * blockDim.x + threadIdx.x;   /* 0..65535 */
    const float m     = fdec(*gmin);
    const float total = supercum[NUM_SUPER];

    _Float16 th[16];
    _Float16 tl[16];

    #pragma unroll 1
    for (int q = 0; q < 16; ++q) {
        const unsigned sid = tid * 16u + (unsigned)q;
        float r = rng_uniform(sid) * total;

        /* level 0: 255 superblocks (inclusive cumsum), 8-step binary search */
        int lo = 0, hi = NUM_SUPER - 1;
        while (lo < hi) {
            const int mid = (lo + hi) >> 1;
            if (r < supercum[mid]) hi = mid; else lo = mid + 1;
        }
        const int s = lo;
        r -= (s ? supercum[s - 1] : 0.0f);

        /* level 1: 512 leaves inside superblock, 9-step binary search */
        const float* lc = leafcum + s * LEAVES_PER_SUP;
        lo = 0; hi = LEAVES_PER_SUP - 1;
        while (lo < hi) {
            const int mid = (lo + hi) >> 1;
            if (r < lc[mid]) hi = mid; else lo = mid + 1;
        }
        const int leaf = lo;
        r -= (leaf ? lc[leaf - 1] : 0.0f);

        /* level 2: sequential scan of one 128-element leaf (L2-resident) */
        const float* base = out + NUM_POS + (size_t)(s * LEAVES_PER_SUP + leaf) * LEAF;
        float v = base[LEAF - 1];
        float acc = 0.0f;
        bool found = false;
        for (int e = 0; e < LEAF && !found; e += 4) {
            const float4 x = *(const float4*)(base + e);
            float c0 = x.x, c1 = x.y, c2 = x.z, c3 = x.w;
            acc += c0 - m; if (acc > r) { v = c0; found = true; }
            if (!found) { acc += c1 - m; if (acc > r) { v = c1; found = true; } }
            if (!found) { acc += c2 - m; if (acc > r) { v = c2; found = true; } }
            if (!found) { acc += c3 - m; if (acc > r) { v = c3; found = true; } }
        }

        /* loss term: log_sigmoid(pos[i] - neg_sample), i = sid mod 65536 */
        const float p = out[sid & (NUM_POS - 1)];
        const float d = p - v;
        const float t = fminf(d, 0.0f) - log1pf(expf(-fabsf(d)));

        const _Float16 h = (_Float16)t;
        th[q] = h;
        tl[q] = (_Float16)(t - (float)h);
    }

    /* pack terms into B operands; A = ones -> column-sum reduction */
    v16h ones, bhi, blo;
    #pragma unroll
    for (int q = 0; q < 16; ++q) {
        ones[q] = (_Float16)1.0f;
        bhi[q]  = th[q];
        blo[q]  = tl[q];
    }
    v8f c = {};
    c = __builtin_amdgcn_wmma_f32_16x16x32_f16(false, ones, false, bhi,
                                               (short)0, c, false, false);
    c = __builtin_amdgcn_wmma_f32_16x16x32_f16(false, ones, false, blo,
                                               (short)0, c, false, false);
    float sgot = c[0];
    #pragma unroll
    for (int off = 16; off; off >>= 1)
        sgot += __shfl_xor(sgot, off, 32);
    sgot *= 0.5f;                                  /* rows replicated twice in c[0] */
    if ((threadIdx.x & 31) == 0)
        partial[tid >> 5] = sgot;
}

/* =====================================================================
 * Pass 4: deterministic final reduction of 2048 wave partials.
 * ===================================================================== */
__global__ __launch_bounds__(256)
void bpr_finalize(const float* __restrict__ partial, float* __restrict__ outv) {
    __shared__ float sm[256];
    float a = 0.0f;
    for (int j = (int)threadIdx.x; j < 2048; j += 256) a += partial[j];
    sm[threadIdx.x] = a;
    __syncthreads();
    for (int off = 128; off; off >>= 1) {
        if ((int)threadIdx.x < off) sm[threadIdx.x] += sm[threadIdx.x + off];
        __syncthreads();
    }
    if (threadIdx.x == 0)
        outv[0] = -(sm[0] / (float)NUM_SAMPLES);
}

extern "C" void kernel_launch(void* const* d_in, const int* in_sizes, int n_in,
                              void* d_out, int out_size, void* d_ws, size_t ws_size,
                              hipStream_t stream) {
    (void)in_sizes; (void)n_in; (void)out_size;
    if (ws_size < (size_t)WS_NEEDED) return;

    const float* output = (const float*)d_in[0];   /* d_in[1] = label (fixed layout) */
    float* outv = (float*)d_out;

    char* ws = (char*)d_ws;
    unsigned* d_min      = (unsigned*)(ws + WS_MIN);
    float*    d_supercum = (float*)(ws + WS_SUPERCUM);
    float*    d_leafsum  = (float*)(ws + WS_LEAFSUM);
    float*    d_leafcum  = (float*)(ws + WS_LEAFCUM);
    float*    d_partial  = (float*)(ws + WS_PARTIAL);

    /* init min cell to +inf encoding (0xFFFFFFFF); graph-capture safe */
    hipMemsetAsync(d_min, 0xFF, sizeof(unsigned), stream);

    bpr_pass1<<<510, 256, 0, stream>>>(output, d_min, d_leafsum);
    bpr_scan<<<1, 256, 0, stream>>>(d_min, d_leafsum, d_leafcum, d_supercum);
    bpr_sample_loss<<<256, 256, 0, stream>>>(output, d_min, d_supercum,
                                             d_leafcum, d_partial);
    bpr_finalize<<<1, 256, 0, stream>>>(d_partial, outv);
}
